// LocalAttention_58652073394312
// MI455X (gfx1250) — compile-verified, matrix operand order checked by reading, not run on hardware
//
#include <hip/hip_runtime.h>
#include <hip/hip_bf16.h>

// ---------------------------------------------------------------------------
// Local (banded) multi-head attention for MI455X (gfx1250), bf16 WMMA pipeline
//   B=8, L=2048, D=512, H=8, dh=64, WINDOW=64
// ---------------------------------------------------------------------------

#define BATCH   8
#define SEQLEN  2048
#define DMODEL  512
#define NHEAD   8
#define DHEAD   64
#define WINDOW  64
#define NTOK    (BATCH * SEQLEN)           // 16384
#define QKVDIM  (3 * DMODEL)               // 1536

typedef __attribute__((ext_vector_type(16))) __bf16          v16bf;
typedef __attribute__((ext_vector_type(16))) unsigned short  v16u;
typedef __attribute__((ext_vector_type(8)))  unsigned short  v8u;   // 16 bytes
typedef __attribute__((ext_vector_type(8)))  float           v8f;

// fp32 -> bf16 bits, round-to-nearest-even
__device__ __forceinline__ unsigned short f2bf_bits(float f) {
  unsigned u = __builtin_bit_cast(unsigned, f);
  unsigned r = u + 0x7FFFu + ((u >> 16) & 1u);
  return (unsigned short)(r >> 16);
}

__device__ __forceinline__ v8f wmma_bf16(v16bf a, v16bf b, v8f c) {
  return __builtin_amdgcn_wmma_f32_16x16x32_bf16(
      /*neg_a=*/false, a, /*neg_b=*/false, b,
      /*c_mod=*/(short)0, c, /*reuse_a=*/false, /*reuse_b=*/false);
}

// Load a 16x32 bf16 A/B operand tile from a row-major matrix.
// For the A operand, "row" = M (0..15); for the B operand, "row" = N (0..15)
// and the source matrix must hold B[k][n] at row n, col k (i.e. K-major rows).
// ISA layout (16-bit A 16x32): lanes 0-15 hold row=lane, K in {0..7, 16..23};
// lanes 16-31 hold row=lane-16, K in {8..15, 24..31}.
__device__ __forceinline__ v16bf load_ab_tile(const unsigned short* base, int ld) {
  int l    = threadIdx.x & 31;
  int half = l >> 4;
  const unsigned short* row = base + (l & 15) * ld + half * 8;
  v16u r;
#pragma unroll
  for (int e = 0; e < 8; ++e) r[e] = row[e];           // K = half*8 + e
#pragma unroll
  for (int e = 0; e < 8; ++e) r[8 + e] = row[16 + e];  // K = 16 + half*8 + e
  return __builtin_bit_cast(v16bf, r);
}

// ---------------------------------------------------------------------------
// Block-cooperative GEMM core: 256 threads (8 waves) compute a 128x64 tile of
// C = A[128xK] * W^T (W row-major [N][K]). Each wave owns 16 rows; all waves
// share the 64-column weight group, staged per K-step into LDS (double
// buffered, 80B-padded rows => conflict-free b128 operand reads).
//   sb: LDS [2][64][40] ushort. Requires K % 32 == 0.
// ---------------------------------------------------------------------------
__device__ __forceinline__ void gemm_block_16x64(
    const unsigned short* __restrict__ Arow,   // A + row0_wave*lda (per wave)
    int lda,
    const unsigned short* __restrict__ Wcol,   // W + col0*ldw (per block)
    int ldw,
    int K,
    unsigned short (*sb)[64][40],
    v8f acc[4]) {
  const int t    = threadIdx.x;
  const int l    = t & 31;
  const int half = l >> 4;
  const int srow = t >> 2;   // 0..63: weight row (output column) staged
  const int sseg = t & 3;    // 16-byte segment within the 32-element K-step
  const int NSTEP = K / 32;

  // stage step 0 into buffer 0
  *(v8u*)&sb[0][srow][sseg * 8] =
      *(const v8u*)(Wcol + (size_t)srow * ldw + sseg * 8);
  v16bf a_cur = load_ab_tile(Arow, lda);
  __syncthreads();

#pragma unroll 2
  for (int s = 0; s < NSTEP; ++s) {
    const int cur = s & 1;
    const bool more = (s + 1 < NSTEP);

    // issue next-step global loads early (B segment + A tile)
    v8u gnext = {};
    if (more)
      gnext = *(const v8u*)(Wcol + (size_t)srow * ldw + (s + 1) * 32 + sseg * 8);
    v16bf a_nxt = a_cur;
    if (more) a_nxt = load_ab_tile(Arow + (s + 1) * 32, lda);

    // B operands from LDS + WMMA
#pragma unroll
    for (int j = 0; j < 4; ++j) {
      int row = j * 16 + (l & 15);
      v8u lo = *(const v8u*)&sb[cur][row][half * 8];
      v8u hi = *(const v8u*)&sb[cur][row][16 + half * 8];
      v16u raw;
#pragma unroll
      for (int e = 0; e < 8; ++e) { raw[e] = lo[e]; raw[8 + e] = hi[e]; }
      acc[j] = wmma_bf16(a_cur, __builtin_bit_cast(v16bf, raw), acc[j]);
    }

    // land next-step B into the other buffer, flip
    if (more) *(v8u*)&sb[cur ^ 1][srow][sseg * 8] = gnext;
    a_cur = a_nxt;
    __syncthreads();
  }
}

// ---------------------------------------------------------------------------
// fp32 -> bf16 conversion
// ---------------------------------------------------------------------------
__global__ __launch_bounds__(256) void k_f32_to_bf16(const float* __restrict__ in,
                                                     unsigned short* __restrict__ out,
                                                     int n) {
  int i = blockIdx.x * blockDim.x + threadIdx.x;
  if (i < n) out[i] = f2bf_bits(in[i]);
}

// ---------------------------------------------------------------------------
// GEMM1: qkv[NTOK x 1536] = x[NTOK x 512] @ W_in^T + b_in ; Q cols pre-scaled
// Block computes 128x64. grid = (16384/128) * (1536/64) = 128*24 = 3072.
// ---------------------------------------------------------------------------
__global__ __launch_bounds__(256) void k_gemm_qkv(
    const unsigned short* __restrict__ Xb,
    const unsigned short* __restrict__ Wb,   // [1536][512]
    const float* __restrict__ bias,          // [1536]
    unsigned short* __restrict__ Qkv) {      // [NTOK][1536]
  __shared__ unsigned short s_b[2][64][40];

  const int NG   = QKVDIM / 64;  // 24
  int ng   = blockIdx.x % NG;
  int msup = blockIdx.x / NG;
  int wave = threadIdx.x >> 5;
  int row0 = msup * 128 + wave * 16;
  int col0 = ng * 64;

  v8f acc[4] = {};
  gemm_block_16x64(Xb + (size_t)row0 * DMODEL, DMODEL,
                   Wb + (size_t)col0 * DMODEL, DMODEL, DMODEL, s_b, acc);

  int l = threadIdx.x & 31;
  int n = l & 15, half = l >> 4;
#pragma unroll
  for (int j = 0; j < 4; ++j) {
    int col = col0 + j * 16 + n;
    float s = (col < DMODEL) ? 0.125f : 1.0f;  // 1/sqrt(dh) pre-applied to Q
    float bv = bias[col];
#pragma unroll
    for (int r = 0; r < 8; ++r) {
      int m = r + 8 * half;
      Qkv[(size_t)(row0 + m) * QKVDIM + col] = f2bf_bits((acc[j][r] + bv) * s);
    }
  }
}

// ---------------------------------------------------------------------------
// Banded attention: one wave per (batch, head, 16-query block).
// grid = 8*8*128 = 8192 blocks of 32 threads.
//
// Score K-tiles are loaded & WMMA'd unconditionally, refilled in place one
// key block ahead. Out-of-range key blocks read garbage that stays inside the
// workspace allocation; their scores are overwritten with -1e30 by the mask,
// so any NaN from garbage bf16 bits is discarded before softmax. V staging
// stays conditional (zeros for out-of-range blocks) since P*garbage may NaN.
// ---------------------------------------------------------------------------
__global__ __launch_bounds__(32) void k_attention(
    const unsigned short* __restrict__ Qkv,
    unsigned short* __restrict__ Attn) {     // [NTOK][512]
  __shared__ float          s_sc[16][160];   // scores (fp32), 144 used + pad
  __shared__ unsigned short s_p[16][160];    // probabilities (bf16)
  __shared__ unsigned short s_v[160][64];    // V block (bf16), 144 used + pad

  int l   = threadIdx.x;           // 0..31
  int idx = blockIdx.x;
  int qb  = idx & 127;
  int h   = (idx >> 7) & 7;
  int b   = idx >> 10;
  int q0  = qb * 16;
  int tb  = b * SEQLEN;

  int n = l & 15, half = l >> 4;
  int vr = l >> 1, vc = (l & 1) * 32;

  // Q operand tiles (16 x 64 split into two 16x32), Q already scaled by 1/8
  const unsigned short* qptr = Qkv + (size_t)(tb + q0) * QKVDIM + h * DHEAD;
  v16bf aq0 = load_ab_tile(qptr, QKVDIM);
  v16bf aq1 = load_ab_tile(qptr + 32, QKVDIM);

  // ---- scores: S = Q K^T over 9 aligned key blocks, masked to |i-j|<=64 ----
  const unsigned short* kbase =
      Qkv + (size_t)(tb + q0 - WINDOW) * QKVDIM + DMODEL + h * DHEAD;
  v16bf bk0 = load_ab_tile(kbase, QKVDIM);
  v16bf bk1 = load_ab_tile(kbase + 32, QKVDIM);

  for (int kb = 0; kb < 9; ++kb) {
    int key0 = q0 - WINDOW + kb * 16;

    v8f c = {};
    c = wmma_bf16(aq0, bk0, c);
    c = wmma_bf16(aq1, bk1, c);

    // refill the K-tile slots in place for the next key block
    if (kb < 8) {
      const unsigned short* knxt = kbase + (size_t)(kb + 1) * 16 * QKVDIM;
      bk0 = load_ab_tile(knxt, QKVDIM);
      bk1 = load_ab_tile(knxt + 32, QKVDIM);
    }

#pragma unroll
    for (int r = 0; r < 8; ++r) {
      int m = r + 8 * half;
      int i = q0 + m, j = key0 + n;
      float v = c[r];
      if (j < 0 || j >= SEQLEN || j < i - WINDOW || j > i + WINDOW) v = -1e30f;
      s_sc[m][kb * 16 + n] = v;
    }

    // stage V rows for this key block (conditional: garbage could be NaN)
    if (key0 >= 0 && key0 < SEQLEN) {
      const unsigned short* vptr =
          Qkv + (size_t)(tb + key0) * QKVDIM + 2 * DMODEL + h * DHEAD;
#pragma unroll
      for (int e = 0; e < 32; ++e)
        s_v[kb * 16 + vr][vc + e] = vptr[(size_t)vr * QKVDIM + vc + e];
    } else {
#pragma unroll
      for (int e = 0; e < 32; ++e) s_v[kb * 16 + vr][vc + e] = 0;
    }
  }
  // pad columns/rows 144..159
#pragma unroll
  for (int r = 0; r < 8; ++r) s_sc[r + 8 * half][144 + n] = -1e30f;
#pragma unroll
  for (int e = 0; e < 32; ++e) s_v[144 + vr][vc + e] = 0;
  __syncthreads();

  // ---- softmax: lanes 0..15 each own one query row ----
  if (l < 16) {
    float mx = -1e30f;
    for (int c = 0; c < 160; ++c) mx = fmaxf(mx, s_sc[l][c]);
    float sum = 0.f;
    for (int c = 0; c < 160; ++c) {
      float e = __expf(s_sc[l][c] - mx);
      s_sc[l][c] = e;
      sum += e;
    }
    float inv = 1.0f / sum;
    for (int c = 0; c < 160; ++c) s_p[l][c] = f2bf_bits(s_sc[l][c] * inv);
  }
  __syncthreads();

  // ---- PV: out[16x64] = P[16x160] * V[160x64] (pad rows are exact zeros) ----
  v8f acc[4] = {};
  for (int kt = 0; kt < 5; ++kt) {
    v16bf a = load_ab_tile(&s_p[0][0] + kt * 32, 160);
#pragma unroll
    for (int j = 0; j < 4; ++j) {
      v16u raw;
#pragma unroll
      for (int e = 0; e < 8; ++e)
        raw[e] = s_v[kt * 32 + half * 8 + e][j * 16 + n];
#pragma unroll
      for (int e = 0; e < 8; ++e)
        raw[8 + e] = s_v[kt * 32 + 16 + half * 8 + e][j * 16 + n];
      v16bf bt = __builtin_bit_cast(v16bf, raw);
      acc[j] = wmma_bf16(a, bt, acc[j]);
    }
  }

  // store merged-head attention output (row = token, col = h*64 + c)
#pragma unroll
  for (int j = 0; j < 4; ++j)
#pragma unroll
    for (int r = 0; r < 8; ++r) {
      int m = r + 8 * half;
      Attn[(size_t)(tb + q0 + m) * DMODEL + h * DHEAD + j * 16 + n] =
          f2bf_bits(acc[j][r]);
    }
}

// ---------------------------------------------------------------------------
// GEMM2: out[NTOK x 512] = attn[NTOK x 512] @ W_out^T + b_out  (fp32 result)
// Block computes 128x64. grid = (16384/128) * (512/64) = 128*8 = 1024.
// ---------------------------------------------------------------------------
__global__ __launch_bounds__(256) void k_gemm_out(
    const unsigned short* __restrict__ Ab,
    const unsigned short* __restrict__ Wb,   // [512][512]
    const float* __restrict__ bias,          // [512]
    float* __restrict__ Out) {               // [NTOK][512]
  __shared__ unsigned short s_b[2][64][40];

  const int NG = DMODEL / 64;  // 8
  int ng   = blockIdx.x % NG;
  int msup = blockIdx.x / NG;
  int wave = threadIdx.x >> 5;
  int row0 = msup * 128 + wave * 16;
  int col0 = ng * 64;

  v8f acc[4] = {};
  gemm_block_16x64(Ab + (size_t)row0 * DMODEL, DMODEL,
                   Wb + (size_t)col0 * DMODEL, DMODEL, DMODEL, s_b, acc);

  int l = threadIdx.x & 31;
  int n = l & 15, half = l >> 4;
#pragma unroll
  for (int j = 0; j < 4; ++j) {
    int col = col0 + j * 16 + n;
    float bv = bias[col];
#pragma unroll
    for (int r = 0; r < 8; ++r) {
      int m = r + 8 * half;
      Out[(size_t)(row0 + m) * DMODEL + col] = acc[j][r] + bv;
    }
  }
}

// ---------------------------------------------------------------------------
extern "C" void kernel_launch(void* const* d_in, const int* in_sizes, int n_in,
                              void* d_out, int out_size, void* d_ws, size_t ws_size,
                              hipStream_t stream) {
  const float* x     = (const float*)d_in[0];  // [8,2048,512]
  const float* in_w  = (const float*)d_in[1];  // [1536,512]
  const float* in_b  = (const float*)d_in[2];  // [1536]
  const float* out_w = (const float*)d_in[3];  // [512,512]
  const float* out_b = (const float*)d_in[4];  // [512]
  float* out = (float*)d_out;                  // [8,2048,512]

  unsigned short* x_bf   = (unsigned short*)d_ws;
  unsigned short* wq_bf  = x_bf   + (size_t)NTOK * DMODEL;       // 1536*512
  unsigned short* wo_bf  = wq_bf  + (size_t)QKVDIM * DMODEL;     // 512*512
  unsigned short* qkv_bf = wo_bf  + (size_t)DMODEL * DMODEL;     // NTOK*1536
  unsigned short* at_bf  = qkv_bf + (size_t)NTOK * QKVDIM;       // NTOK*512

  // 1) fp32 -> bf16 staging
  {
    int n = NTOK * DMODEL;
    k_f32_to_bf16<<<(n + 255) / 256, 256, 0, stream>>>(x, x_bf, n);
  }
  {
    int n = QKVDIM * DMODEL;
    k_f32_to_bf16<<<(n + 255) / 256, 256, 0, stream>>>(in_w, wq_bf, n);
  }
  {
    int n = DMODEL * DMODEL;
    k_f32_to_bf16<<<(n + 255) / 256, 256, 0, stream>>>(out_w, wo_bf, n);
  }

  // 2) QKV projection (Q pre-scaled by 1/sqrt(dh))
  k_gemm_qkv<<<(NTOK / 128) * (QKVDIM / 64), 256, 0, stream>>>(
      x_bf, wq_bf, in_b, qkv_bf);

  // 3) banded attention
  k_attention<<<BATCH * NHEAD * (SEQLEN / 16), 32, 0, stream>>>(qkv_bf, at_bf);

  // 4) output projection
  k_gemm_out<<<(NTOK / 128) * (DMODEL / 64), 256, 0, stream>>>(
      at_bf, wo_bf, out_b, out);
}
